// AttentionWithLoRA_48765058679462
// MI455X (gfx1250) — matrix operand order repaired and checked by reading, not compiled, not run on hardware
//
#include <hip/hip_runtime.h>
#include <hip/hip_bf16.h>

// ---------------------------------------------------------------------------
// AttentionWithLoRA for MI455X (gfx1250): f16 WMMA everywhere, f32 accumulate.
// LoRA folded into effective weights: x@W.T + s*(x@A.T)@B.T = x@(W+s*B@A).T
// ---------------------------------------------------------------------------

typedef __attribute__((ext_vector_type(16))) _Float16 v16h;
typedef __attribute__((ext_vector_type(8)))  _Float16 v8h;
typedef __attribute__((ext_vector_type(4)))  _Float16 v4h;
typedef __attribute__((ext_vector_type(8)))  float    v8f;

#define DIMN 128
#define NH 4
#define HD 32
#define SEQ 256
#define BATCH 128
#define RANKN 8
#define LORA_SCALE 2.0f
#define INV_SQRT_HD 0.17677669529663687f
#define MROWS (BATCH*SEQ)            // 32768
#define SC_STRIDE 260                 // padded f32 score row stride (dwords)
#define P_STRIDE  264                 // padded f16 P row stride (halves)
#define ATTN_WAVES 2
#define ATTN_LDS_BYTES (ATTN_WAVES*(16*SC_STRIDE*4 + 16*P_STRIDE*2))  // 50176

__device__ __forceinline__ v16h pack16(v8h lo, v8h hi) {
  return __builtin_shufflevector(lo, hi, 0,1,2,3,4,5,6,7,8,9,10,11,12,13,14,15);
}
__device__ __forceinline__ v8f vzero8() {
  v8f z;
  #pragma unroll
  for (int i = 0; i < 8; ++i) z[i] = 0.0f;
  return z;
}
__device__ __forceinline__ v8f wmma_f16(v16h a, v16h b, v8f c) {
  // D = A(16x32 f16) * B(32x16 f16) + C(16x16 f32)
  return __builtin_amdgcn_wmma_f32_16x16x32_f16(false, a, false, b, (short)0, c,
                                                false, false);
}

// ---------------------------------------------------------------------------
// Kernel 1: effective weights (W + 2*B@A) -> f16, wo -> f16, RoPE cos/sin table
// ---------------------------------------------------------------------------
__global__ void lorafuse_prep_kernel(
    const float* __restrict__ wq, const float* __restrict__ wk,
    const float* __restrict__ wv, const float* __restrict__ wo,
    const float* __restrict__ Aq, const float* __restrict__ Bq,
    const float* __restrict__ Ak, const float* __restrict__ Bk,
    const float* __restrict__ Av, const float* __restrict__ Bv,
    _Float16* __restrict__ weff,     // 3 * 128*128, [q|k|v]
    _Float16* __restrict__ wo16,     // 128*128
    float2*   __restrict__ rope)     // [SEQ][HD/2] (cos, sin)
{
  int idx = blockIdx.x * 256 + threadIdx.x;
  if (idx < 3 * DIMN * DIMN) {
    int mat = idx / (DIMN * DIMN);
    int rem = idx % (DIMN * DIMN);
    int n = rem / DIMN, k = rem % DIMN;
    const float* w  = (mat == 0) ? wq : (mat == 1) ? wk : wv;
    const float* A  = (mat == 0) ? Aq : (mat == 1) ? Ak : Av;
    const float* Bm = (mat == 0) ? Bq : (mat == 1) ? Bk : Bv;
    float l = 0.0f;
    #pragma unroll
    for (int r = 0; r < RANKN; ++r) l += Bm[n * RANKN + r] * A[r * DIMN + k];
    weff[idx] = (_Float16)(w[n * DIMN + k] + LORA_SCALE * l);
  } else if (idx < 4 * DIMN * DIMN) {
    int i = idx - 3 * DIMN * DIMN;
    wo16[i] = (_Float16)wo[i];
  } else if (idx < 4 * DIMN * DIMN + SEQ * (HD / 2)) {
    int i = idx - 4 * DIMN * DIMN;
    int s = i / (HD / 2), j = i % (HD / 2);
    float inv_freq = __powf(10000.0f, -(float)(2 * j) / (float)HD);
    float f = (float)s * inv_freq;
    rope[i] = make_float2(__cosf(f), __sinf(f));
  }
}

// ---------------------------------------------------------------------------
// Kernel 2: x f32 -> f16
// ---------------------------------------------------------------------------
__global__ void lorafuse_cvt_kernel(const float* __restrict__ x,
                                    _Float16* __restrict__ x16, int n4)
{
  int i = blockIdx.x * 256 + threadIdx.x;
  if (i < n4) {
    float4 v = ((const float4*)x)[i];
    v4h o;
    o[0] = (_Float16)v.x; o[1] = (_Float16)v.y;
    o[2] = (_Float16)v.z; o[3] = (_Float16)v.w;
    *(v4h*)(x16 + 4 * i) = o;
  }
}

// ---------------------------------------------------------------------------
// Kernel 3: QKV projection GEMM (per-wave 16x128 strip) + RoPE + layout
//   z=0: q -> [b*NH+h][s][32] (rope)   z=1: k -> same (rope)
//   z=2: v -> [b*NH+h][32][s] (transposed for the P@V B-operand)
// ---------------------------------------------------------------------------
__global__ void __launch_bounds__(128)
lorafuse_proj_kernel(const _Float16* __restrict__ x16,
                     const _Float16* __restrict__ weff,
                     const float2*   __restrict__ rope,
                     _Float16* __restrict__ q_ws,
                     _Float16* __restrict__ k_ws,
                     _Float16* __restrict__ v_t)
{
  const int lane = threadIdx.x & 31;
  const int wid  = threadIdx.x >> 5;
  const int z    = blockIdx.y;
  const int m_base = (blockIdx.x * 4 + wid) * 16;
  const int ln = lane & 15, hl = lane >> 4;

  const _Float16* W    = weff + z * DIMN * DIMN;
  const _Float16* arow = x16 + (size_t)(m_base + ln) * DIMN;

  v8f acc[8];
  #pragma unroll
  for (int t = 0; t < 8; ++t) acc[t] = vzero8();

  #pragma unroll
  for (int kc = 0; kc < DIMN; kc += 32) {
    v16h a = pack16(*(const v8h*)(arow + kc + 16 * hl),
                    *(const v8h*)(arow + kc + 16 * hl + 8));
    #pragma unroll
    for (int nt = 0; nt < 8; ++nt) {
      const _Float16* brow = W + (nt * 16 + ln) * DIMN + kc + 16 * hl;
      v16h b = pack16(*(const v8h*)brow, *(const v8h*)(brow + 8));
      acc[nt] = wmma_f16(a, b, acc[nt]);
    }
  }

  #pragma unroll
  for (int nt = 0; nt < 8; ++nt) {
    const int n = nt * 16 + ln;
    const int head = n >> 5;
    const int d = n & (HD - 1);
    #pragma unroll
    for (int r = 0; r < 8; ++r) {
      const int m = m_base + r + 8 * hl;
      const int b = m >> 8;          // / SEQ
      const int s = m & (SEQ - 1);
      float val = acc[nt][r];
      if (z < 2) {
        // RoPE: even/odd d pair lives in adjacent lanes (N = lane&15)
        float2 cs = rope[s * (HD / 2) + (d >> 1)];
        float other = __shfl_xor(val, 1, 32);
        val = (d & 1) ? (other * cs.y + val * cs.x)
                      : (val * cs.x - other * cs.y);
        _Float16* dst = (z == 0) ? q_ws : k_ws;
        dst[((size_t)(b * NH + head) * SEQ + s) * HD + d] = (_Float16)val;
      } else {
        v_t[((size_t)(b * NH + head) * HD + d) * SEQ + s] = (_Float16)val;
      }
    }
  }
}

// ---------------------------------------------------------------------------
// Kernel 4: causal attention, one wave per (b,h,16-row query tile).
//   pass1: scores = scale * q k^T   (1 WMMA per 16x16 tile, only kt<=qt)
//   pass2: softmax in LDS (rowmax/rowsum via shfl_xor 16)
//   pass3: out = P @ V (WMMA, A from LDS, B = v^T from global)
// ---------------------------------------------------------------------------
__global__ void __launch_bounds__(ATTN_WAVES * 32)
lorafuse_attn_kernel(const _Float16* __restrict__ q_ws,
                     const _Float16* __restrict__ k_ws,
                     const _Float16* __restrict__ v_t,
                     _Float16* __restrict__ o_ws)
{
  extern __shared__ char smem[];
  const int lane = threadIdx.x & 31;
  const int wid  = threadIdx.x >> 5;
  const int gt = blockIdx.x * ATTN_WAVES + wid;   // 0..8191
  const int bh = gt >> 4;                          // 0..511
  const int qt = gt & 15;                          // query tile 0..15
  const int ln = lane & 15, hl = lane >> 4;

  float*    sc = (float*)smem + wid * (16 * SC_STRIDE);
  _Float16* pl = (_Float16*)(smem + ATTN_WAVES * 16 * SC_STRIDE * 4)
                 + wid * (16 * P_STRIDE);

  const _Float16* qb = q_ws + (size_t)bh * SEQ * HD;
  const _Float16* kb = k_ws + (size_t)bh * SEQ * HD;
  const _Float16* vb = v_t  + (size_t)bh * HD * SEQ;

  // A operand: 16 query rows, K = HD = 32 (one WMMA K-step)
  const _Float16* ar = qb + (qt * 16 + ln) * HD;
  const v16h a = pack16(*(const v8h*)(ar + 16 * hl),
                        *(const v8h*)(ar + 16 * hl + 8));

  for (int kt = 0; kt <= qt; ++kt) {
    const _Float16* br = kb + (kt * 16 + ln) * HD + 16 * hl;
    v16h b = pack16(*(const v8h*)br, *(const v8h*)(br + 8));
    v8f s8 = wmma_f16(a, b, vzero8());
    const int n = kt * 16 + ln;                    // key column in [0,256)
    #pragma unroll
    for (int r = 0; r < 8; ++r) {
      const int mrow = r + 8 * hl;                 // local query row 0..15
      float v = s8[r] * INV_SQRT_HD;
      if (n > qt * 16 + mrow) v = -3.0e38f;        // causal mask (diag tile)
      sc[mrow * SC_STRIDE + n] = v;
    }
  }
  __syncthreads();

  // ---- softmax over valid columns [0, kmax) ----
  const int kmax = (qt + 1) * 16;
  const int row = ln;
  float* srow = sc + row * SC_STRIDE + hl * 128;
  int cnt = kmax - hl * 128;
  cnt = cnt < 0 ? 0 : (cnt > 128 ? 128 : cnt);
  float mx = -3.4e38f;
  for (int t = 0; t < cnt; ++t) mx = fmaxf(mx, srow[t]);
  mx = fmaxf(mx, __shfl_xor(mx, 16, 32));
  float sum = 0.0f;
  _Float16* prow = pl + row * P_STRIDE + hl * 128;
  for (int t = 0; t < 128; ++t) {
    float e = (t < cnt) ? __expf(srow[t] - mx) : 0.0f;
    sum += e;
    prow[t] = (_Float16)e;                         // zero-fill beyond kmax
  }
  sum += __shfl_xor(sum, 16, 32);
  const float invsum = 1.0f / sum;                 // lane (row) & (row+16)
  __syncthreads();

  // ---- out = P @ V  (K chunks of 32, only valid region) ----
  v8f acc0 = vzero8(), acc1 = vzero8();
  const int nk = (kmax + 31) >> 5;
  for (int c = 0; c < nk; ++c) {
    const int kc = c * 32;
    v16h a2 = pack16(*(const v8h*)(pl + ln * P_STRIDE + kc + 16 * hl),
                     *(const v8h*)(pl + ln * P_STRIDE + kc + 16 * hl + 8));
    const _Float16* vp0 = vb + ln * SEQ + kc + 16 * hl;          // d = ln
    acc0 = wmma_f16(a2, pack16(*(const v8h*)vp0, *(const v8h*)(vp0 + 8)), acc0);
    const _Float16* vp1 = vb + (16 + ln) * SEQ + kc + 16 * hl;   // d = 16+ln
    acc1 = wmma_f16(a2, pack16(*(const v8h*)vp1, *(const v8h*)(vp1 + 8)), acc1);
  }

  const int b = bh >> 2, head = bh & (NH - 1);
  #pragma unroll
  for (int r = 0; r < 8; ++r) {
    const int m = r + 8 * hl;
    const float iv = __shfl(invsum, m, 32);        // lane m holds row m's 1/sum
    const size_t grow = ((size_t)(b * SEQ + qt * 16 + m)) * DIMN + head * HD;
    o_ws[grow + ln]      = (_Float16)(acc0[r] * iv);
    o_ws[grow + 16 + ln] = (_Float16)(acc1[r] * iv);
  }
}

// ---------------------------------------------------------------------------
// Kernel 5: final projection out = attn_out @ wo.T  (f32 output)
// ---------------------------------------------------------------------------
__global__ void __launch_bounds__(128)
lorafuse_outproj_kernel(const _Float16* __restrict__ o16,
                        const _Float16* __restrict__ wo16,
                        float* __restrict__ out)
{
  const int lane = threadIdx.x & 31;
  const int wid  = threadIdx.x >> 5;
  const int m_base = (blockIdx.x * 4 + wid) * 16;
  const int ln = lane & 15, hl = lane >> 4;
  const _Float16* arow = o16 + (size_t)(m_base + ln) * DIMN;

  v8f acc[8];
  #pragma unroll
  for (int t = 0; t < 8; ++t) acc[t] = vzero8();

  #pragma unroll
  for (int kc = 0; kc < DIMN; kc += 32) {
    v16h a = pack16(*(const v8h*)(arow + kc + 16 * hl),
                    *(const v8h*)(arow + kc + 16 * hl + 8));
    #pragma unroll
    for (int nt = 0; nt < 8; ++nt) {
      const _Float16* brow = wo16 + (nt * 16 + ln) * DIMN + kc + 16 * hl;
      v16h b = pack16(*(const v8h*)brow, *(const v8h*)(brow + 8));
      acc[nt] = wmma_f16(a, b, acc[nt]);
    }
  }
  #pragma unroll
  for (int nt = 0; nt < 8; ++nt) {
    #pragma unroll
    for (int r = 0; r < 8; ++r) {
      out[(size_t)(m_base + r + 8 * hl) * DIMN + nt * 16 + ln] = acc[nt][r];
    }
  }
}

// ---------------------------------------------------------------------------
extern "C" void kernel_launch(void* const* d_in, const int* in_sizes, int n_in,
                              void* d_out, int out_size, void* d_ws, size_t ws_size,
                              hipStream_t stream)
{
  const float* x  = (const float*)d_in[0];
  const float* wq = (const float*)d_in[1];
  const float* wk = (const float*)d_in[2];
  const float* wv = (const float*)d_in[3];
  const float* wo = (const float*)d_in[4];
  const float* Aq = (const float*)d_in[5];
  const float* Bq = (const float*)d_in[6];
  const float* Ak = (const float*)d_in[7];
  const float* Bk = (const float*)d_in[8];
  const float* Av = (const float*)d_in[9];
  const float* Bv = (const float*)d_in[10];
  float* out = (float*)d_out;

  char* ws = (char*)d_ws;
  size_t off = 0;
  _Float16* x16  = (_Float16*)(ws + off); off += (size_t)MROWS * DIMN * 2;  // 8 MiB
  _Float16* weff = (_Float16*)(ws + off); off += 3 * DIMN * DIMN * 2;       // 96 KiB
  _Float16* wo16 = (_Float16*)(ws + off); off += DIMN * DIMN * 2;           // 32 KiB
  float2*   rope = (float2*)  (ws + off); off += SEQ * (HD / 2) * 8;        // 32 KiB
  _Float16* q_ws = (_Float16*)(ws + off); off += (size_t)BATCH * NH * SEQ * HD * 2;
  _Float16* k_ws = (_Float16*)(ws + off); off += (size_t)BATCH * NH * SEQ * HD * 2;
  _Float16* v_t  = (_Float16*)(ws + off); off += (size_t)BATCH * NH * HD * SEQ * 2;
  _Float16* o_ws = (_Float16*)(ws + off); off += (size_t)MROWS * DIMN * 2;
  (void)ws_size; (void)in_sizes; (void)n_in; (void)out_size;

  // 1) effective weights + rope table
  lorafuse_prep_kernel<<<272, 256, 0, stream>>>(wq, wk, wv, wo, Aq, Bq, Ak, Bk,
                                                Av, Bv, weff, wo16, rope);
  // 2) x -> f16
  lorafuse_cvt_kernel<<<(MROWS * DIMN / 4 + 255) / 256, 256, 0, stream>>>(
      x, x16, MROWS * DIMN / 4);
  // 3) fused QKV projection + RoPE (grid.y selects q/k/v)
  dim3 gproj(MROWS / 16 / 4, 3);
  lorafuse_proj_kernel<<<gproj, 128, 0, stream>>>(x16, weff, rope, q_ws, k_ws, v_t);
  // 4) causal attention
  lorafuse_attn_kernel<<<(BATCH * NH * (SEQ / 16)) / ATTN_WAVES, ATTN_WAVES * 32,
                         ATTN_LDS_BYTES, stream>>>(q_ws, k_ws, v_t, o_ws);
  // 5) output projection (f32)
  lorafuse_outproj_kernel<<<MROWS / 16 / 4, 128, 0, stream>>>(o_ws, wo16, out);
}
